// SceneReasoningLayer_90598040141846
// MI455X (gfx1250) — matrix operand order repaired and checked by reading, not compiled, not run on hardware
//
#include <hip/hip_runtime.h>
#include <hip/hip_bf16.h>
#include <math.h>

typedef __attribute__((ext_vector_type(16))) __bf16 v16bf;
typedef __attribute__((ext_vector_type(8)))  float  v8f;

#define D_MODEL 768
#define NHEAD   8
#define HEADD   96
#define MLPD    3072
#define SEQ     2048
#define BATCH   4
#define NTOK    (SEQ * BATCH)   // 8192

static __device__ __forceinline__ v8f v8zero() {
    v8f z = {0.f, 0.f, 0.f, 0.f, 0.f, 0.f, 0.f, 0.f};
    return z;
}

union FragU {
    uint4 q[2];
    v16bf v;
};

// CDNA5 async DMA: global -> LDS, 16 bytes per lane (tracked by ASYNCcnt)
static __device__ __forceinline__ void async_load_b128(unsigned lds_byte_off,
                                                       unsigned long long gaddr) {
    asm volatile("global_load_async_to_lds_b128 %0, %1, off"
                 :: "v"(lds_byte_off), "v"(gaddr) : "memory");
}
static __device__ __forceinline__ void wait_async_le1() {
    asm volatile("s_wait_asynccnt 0x1" ::: "memory");
}
static __device__ __forceinline__ void wait_async_le0() {
    asm volatile("s_wait_asynccnt 0x0" ::: "memory");
}

// ---------------------------------------------------------------------------
// Weight fp32[K][N] -> bf16 swizzled into WMMA B-fragment order:
// out[((ks*(N/16) + nt)*32 + lane)*16 + e] = in[ks*32 + (lane>>4)*16 + e][nt*16 + (lane&15)]
// so a B fragment (32x16 block) is 32 contiguous bytes per lane.
// ---------------------------------------------------------------------------
__global__ void cvtz_kernel(const float* __restrict__ in,
                            __bf16* __restrict__ out, int K, int N) {
    int i = blockIdx.x * 256 + threadIdx.x;
    if (i >= K * N) return;
    int e  = i & 15;
    int l  = (i >> 4) & 31;
    int r  = i >> 9;
    int nt = r % (N >> 4);
    int bk = r / (N >> 4);
    int k  = bk * 32 + ((l >> 4) << 4) + e;
    int c  = (nt << 4) + (l & 15);
    out[i] = (__bf16)in[(size_t)k * N + c];
}

// ---------------------------------------------------------------------------
// LayerNorm: one block (256 threads) per token row of 768; writes bf16
// ---------------------------------------------------------------------------
__global__ __launch_bounds__(256) void ln_kernel(const float* __restrict__ x,
                                                 const float* __restrict__ g,
                                                 const float* __restrict__ b,
                                                 __bf16* __restrict__ out) {
    int row = blockIdx.x;
    int tid = threadIdx.x;
    const float* xr = x + (size_t)row * D_MODEL;
    float v0 = xr[tid], v1 = xr[tid + 256], v2 = xr[tid + 512];
    __shared__ float rs[256], rss[256];
    rs[tid]  = v0 + v1 + v2;
    rss[tid] = v0 * v0 + v1 * v1 + v2 * v2;
    __syncthreads();
    for (int o = 128; o > 0; o >>= 1) {
        if (tid < o) { rs[tid] += rs[tid + o]; rss[tid] += rss[tid + o]; }
        __syncthreads();
    }
    float mu   = rs[0] * (1.0f / D_MODEL);
    float var  = rss[0] * (1.0f / D_MODEL) - mu * mu;
    float rinv = rsqrtf(var + 1e-5f);
    __bf16* orow = out + (size_t)row * D_MODEL;
    orow[tid]       = (__bf16)((v0 - mu) * rinv * g[tid]       + b[tid]);
    orow[tid + 256] = (__bf16)((v1 - mu) * rinv * g[tid + 256] + b[tid + 256]);
    orow[tid + 512] = (__bf16)((v2 - mu) * rinv * g[tid + 512] + b[tid + 512]);
}

// ---------------------------------------------------------------------------
// bf16 GEMM, block tile 64x64 (8 waves, one 16x32 tile each).
// A tiles: async-DMA to double-buffered LDS (ASYNCcnt pipelined).
// B fragments: 32B contiguous loads from pre-swizzled L2-resident weights,
// software-pipelined one K-step ahead.  ACT: 0 none, 1 exact GELU.
// ---------------------------------------------------------------------------
template <int ACT, bool RESID, bool OUTBF>
__global__ __launch_bounds__(256) void gemm_kernel(
    const __bf16* __restrict__ A, const __bf16* __restrict__ Bz,
    const float* __restrict__ bias, const float* __restrict__ resid,
    void* __restrict__ out, int M, int Nn, int K) {
    constexpr int AP = 40;  // padded LDS row stride (bf16) for A (64x32)
    __shared__ __align__(16) __bf16 Al[2][64 * AP];

    int tid  = threadIdx.x;
    int wid  = tid >> 5, lane = tid & 31;
    int half = lane >> 4, ln16 = lane & 15;
    int wr = wid & 3, wc = wid >> 2;
    int bM = blockIdx.y * 64, bN = blockIdx.x * 64;
    int nk = K >> 5;
    int ntiles = Nn >> 4;

    // per-thread async slot: one b128 per thread covers the 64x32 A tile
    int ar = tid >> 2, ach = tid & 3;
    unsigned long long gA =
        (unsigned long long)(size_t)(A + (size_t)(bM + ar) * K + ach * 8);
    unsigned ldsA0 = (unsigned)(size_t)&Al[0][ar * AP + ach * 8];
    unsigned ldsA1 = (unsigned)(size_t)&Al[1][ar * AP + ach * 8];

    async_load_b128(ldsA0, gA);  // prefetch tile 0

    // B fragment base for this wave (two 16-col tiles)
    const __bf16* Bl0 = Bz + (size_t)((bN >> 4) + wc * 2) * 512 + lane * 16;
    v16bf bn0 = *(const v16bf*)(Bl0);
    v16bf bn1 = *(const v16bf*)(Bl0 + 512);

    v8f acc0 = v8zero(), acc1 = v8zero();

    for (int ks = 0; ks < nk; ks++) {
        if (ks + 1 < nk) {
            async_load_b128((ks & 1) ? ldsA0 : ldsA1,
                            gA + (unsigned long long)(ks + 1) * 64);
            wait_async_le1();
        } else {
            wait_async_le0();
        }
        __syncthreads();

        FragU au;
        const __bf16* arow = &Al[ks & 1][(wr * 16 + ln16) * AP];
        au.q[0] = *(const uint4*)(arow + 16 * half);
        au.q[1] = *(const uint4*)(arow + 16 * half + 8);

        v16bf b0 = bn0, b1 = bn1;
        if (ks + 1 < nk) {  // pipeline next B fragments (L2 hits)
            const __bf16* bp = Bl0 + (size_t)(ks + 1) * ntiles * 512;
            bn0 = *(const v16bf*)(bp);
            bn1 = *(const v16bf*)(bp + 512);
        }
        acc0 = __builtin_amdgcn_wmma_f32_16x16x32_bf16(false, au.v, false, b0,
                                                       (short)0, acc0, false, false);
        acc1 = __builtin_amdgcn_wmma_f32_16x16x32_bf16(false, au.v, false, b1,
                                                       (short)0, acc1, false, false);
        __syncthreads();
    }

    // epilogue: C layout -> row = i + 8*half, col = ln16
#pragma unroll
    for (int s = 0; s < 2; s++) {
        v8f acc = s ? acc1 : acc0;
#pragma unroll
        for (int i = 0; i < 8; i++) {
            int row = bM + wr * 16 + i + 8 * half;
            int col = bN + wc * 32 + s * 16 + ln16;
            float v = acc[i] + bias[col];
            if (ACT == 1) v = 0.5f * v * (1.0f + erff(v * 0.70710678118654752f));
            if (RESID) v += resid[(size_t)row * Nn + col];
            if (OUTBF)
                ((__bf16*)out)[(size_t)row * Nn + col] = (__bf16)v;
            else
                ((float*)out)[(size_t)row * Nn + col] = v;
        }
    }
}

// ---------------------------------------------------------------------------
// Flash attention: grid (B*H, SEQ/128); block = 8 waves, 16 query rows/wave.
// K chunk (32 keys x 96) async-DMA'd into LDS; V staged TRANSPOSED so P.V
// B-fragments are contiguous b128 loads; online softmax per 16-lane half.
// ---------------------------------------------------------------------------
__global__ __launch_bounds__(256) void attn_kernel(
    const __bf16* __restrict__ Q, const __bf16* __restrict__ Kb,
    const __bf16* __restrict__ Vb, const float* __restrict__ bias,
    __bf16* __restrict__ Ob) {
    constexpr int KP = 104;  // K tile row stride (bf16): 208B, 16B-aligned
    constexpr int VP = 40;   // transposed V row stride (32 keys + pad)
    constexpr int PP = 40;   // P scratch row stride
    __shared__ __align__(16) __bf16 Kl[32 * KP];
    __shared__ __align__(16) __bf16 Vt[96 * VP];
    __shared__ __align__(16) __bf16 Pl[8 * 16 * PP];

    int bh = blockIdx.x;
    int b = bh / NHEAD, h = bh % NHEAD;
    int tid = threadIdx.x;
    int wid = tid >> 5, lane = tid & 31;
    int half = lane >> 4, ln16 = lane & 15;
    int q0 = blockIdx.y * 128 + wid * 16;
    const float scale = 0.1020620726f;  // 1/sqrt(96)

    // Q fragments: 16 rows x 96 dims = 3 A-fragments (b128-pair loads)
    v16bf qf[3];
    const __bf16* qrow =
        Q + ((size_t)b * SEQ + q0 + ln16) * D_MODEL + h * HEADD;
#pragma unroll
    for (int t = 0; t < 3; t++) {
        FragU qu;
        qu.q[0] = *(const uint4*)(qrow + 32 * t + 16 * half);
        qu.q[1] = *(const uint4*)(qrow + 32 * t + 16 * half + 8);
        qf[t] = qu.v;
    }

    v8f o[6];
#pragma unroll
    for (int f = 0; f < 6; f++) o[f] = v8zero();
    float mrow[8], lrow[8];
#pragma unroll
    for (int i = 0; i < 8; i++) { mrow[i] = -INFINITY; lrow[i] = 0.f; }

    __bf16* Pw = &Pl[wid * 16 * PP];
    unsigned klds = (unsigned)(size_t)&Kl[0];

    for (int kc = 0; kc < SEQ; kc += 32) {
        __syncthreads();  // all waves done reading previous K/V tiles
        // K tile: async DMA, 384 x 16B chunks
        for (int c = tid; c < 384; c += 256) {
            int r = c / 12, u = c % 12;
            unsigned long long g = (unsigned long long)(size_t)(
                Kb + ((size_t)b * SEQ + kc + r) * D_MODEL + h * HEADD);
            async_load_b128(klds + r * 208 + u * 16, g + u * 16);
        }
        // V tile: load row-major, store transposed (Vt[dim][key])
        for (int t = tid; t < 32 * 48; t += 256) {
            int r = t / 48, u = t % 48;
            unsigned int w = *(const unsigned int*)(
                Vb + ((size_t)b * SEQ + kc + r) * D_MODEL + h * HEADD + u * 2);
            Vt[(2 * u) * VP + r]     = ((const __bf16*)&w)[0];
            Vt[(2 * u + 1) * VP + r] = ((const __bf16*)&w)[1];
        }
        wait_async_le0();
        __syncthreads();

        // S(16x32) = Q . K^T : K fragments are contiguous rows of Kl
        v8f s0 = v8zero(), s1 = v8zero();
#pragma unroll
        for (int t = 0; t < 3; t++) {
            FragU k0u, k1u;
            const __bf16* kr0 = &Kl[ln16 * KP + 32 * t + 16 * half];
            const __bf16* kr1 = &Kl[(16 + ln16) * KP + 32 * t + 16 * half];
            k0u.q[0] = *(const uint4*)(kr0);
            k0u.q[1] = *(const uint4*)(kr0 + 8);
            k1u.q[0] = *(const uint4*)(kr1);
            k1u.q[1] = *(const uint4*)(kr1 + 8);
            s0 = __builtin_amdgcn_wmma_f32_16x16x32_bf16(false, qf[t], false, k0u.v,
                                                         (short)0, s0, false, false);
            s1 = __builtin_amdgcn_wmma_f32_16x16x32_bf16(false, qf[t], false, k1u.v,
                                                         (short)0, s1, false, false);
        }

        // scale + position bias + online softmax (rows live in 16-lane halves)
        float alpha[8];
#pragma unroll
        for (int i = 0; i < 8; i++) {
            int grow = q0 + i + 8 * half;
            s0[i] = s0[i] * scale + bias[(size_t)grow * SEQ + kc + ln16];
            s1[i] = s1[i] * scale + bias[(size_t)grow * SEQ + kc + 16 + ln16];
            float mx = fmaxf(s0[i], s1[i]);
#pragma unroll
            for (int d = 1; d < 16; d <<= 1) mx = fmaxf(mx, __shfl_xor(mx, d, 32));
            float mn = fmaxf(mrow[i], mx);
            alpha[i] = expf(mrow[i] - mn);
            float p0 = expf(s0[i] - mn);
            float p1 = expf(s1[i] - mn);
            float rsum = p0 + p1;
#pragma unroll
            for (int d = 1; d < 16; d <<= 1) rsum += __shfl_xor(rsum, d, 32);
            lrow[i] = lrow[i] * alpha[i] + rsum;
            mrow[i] = mn;
            s0[i] = p0;
            s1[i] = p1;
        }
#pragma unroll
        for (int f = 0; f < 6; f++)
#pragma unroll
            for (int i = 0; i < 8; i++) o[f][i] *= alpha[i];

        // P: C-layout -> per-wave LDS -> A-fragment layout (in-order DS per wave)
#pragma unroll
        for (int i = 0; i < 8; i++) {
            int r = i + 8 * half;
            Pw[r * PP + ln16]      = (__bf16)s0[i];
            Pw[r * PP + 16 + ln16] = (__bf16)s1[i];
        }
        FragU pu;
        const __bf16* pr = &Pw[ln16 * PP];
        pu.q[0] = *(const uint4*)(pr + 16 * half);
        pu.q[1] = *(const uint4*)(pr + 16 * half + 8);

        // O(16x96) += P(16x32) . V(32x96); V fragments contiguous in Vt
#pragma unroll
        for (int f = 0; f < 6; f++) {
            FragU vu;
            const __bf16* vr = &Vt[(f * 16 + ln16) * VP + 16 * half];
            vu.q[0] = *(const uint4*)(vr);
            vu.q[1] = *(const uint4*)(vr + 8);
            o[f] = __builtin_amdgcn_wmma_f32_16x16x32_bf16(false, pu.v, false, vu.v,
                                                           (short)0, o[f], false, false);
        }
    }

    // finalize: divide by row sum, store bf16 [B,N,H*HD]
#pragma unroll
    for (int f = 0; f < 6; f++)
#pragma unroll
        for (int i = 0; i < 8; i++) {
            size_t grow = (size_t)b * SEQ + q0 + i + 8 * half;
            int gcol = h * HEADD + f * 16 + ln16;
            Ob[grow * D_MODEL + gcol] = (__bf16)(o[f][i] / lrow[i]);
        }
}

// ---------------------------------------------------------------------------
// Host orchestration
// ---------------------------------------------------------------------------
extern "C" void kernel_launch(void* const* d_in, const int* in_sizes, int n_in,
                              void* d_out, int out_size, void* d_ws, size_t ws_size,
                              hipStream_t stream) {
    const float* x    = (const float*)d_in[0];
    const float* pb   = (const float*)d_in[1];
    const float* wq   = (const float*)d_in[2];
    const float* bq   = (const float*)d_in[3];
    const float* wk   = (const float*)d_in[4];
    const float* bk   = (const float*)d_in[5];
    const float* wv   = (const float*)d_in[6];
    const float* bv   = (const float*)d_in[7];
    const float* wo   = (const float*)d_in[8];
    const float* bo   = (const float*)d_in[9];
    const float* g1   = (const float*)d_in[10];
    const float* b1   = (const float*)d_in[11];
    const float* g2   = (const float*)d_in[12];
    const float* b2   = (const float*)d_in[13];
    const float* wff1 = (const float*)d_in[14];
    const float* bff1 = (const float*)d_in[15];
    const float* wff2 = (const float*)d_in[16];
    const float* bff2 = (const float*)d_in[17];

    char* ws = (char*)d_ws;
    const size_t SZ_W   = (size_t)D_MODEL * D_MODEL * 2;
    const size_t SZ_WFF = (size_t)D_MODEL * MLPD * 2;
    const size_t SZ_TOK = (size_t)NTOK * D_MODEL * 2;

    __bf16* wq_b   = (__bf16*)(ws);
    __bf16* wk_b   = (__bf16*)(ws + SZ_W);
    __bf16* wv_b   = (__bf16*)(ws + 2 * SZ_W);
    __bf16* wo_b   = (__bf16*)(ws + 3 * SZ_W);
    __bf16* wff1_b = (__bf16*)(ws + 4 * SZ_W);
    __bf16* wff2_b = (__bf16*)(ws + 4 * SZ_W + SZ_WFF);
    char*   base   = ws + 4 * SZ_W + 2 * SZ_WFF;
    __bf16* n1_b   = (__bf16*)(base);
    __bf16* q_b    = (__bf16*)(base + SZ_TOK);
    __bf16* k_b    = (__bf16*)(base + 2 * SZ_TOK);
    __bf16* v_b    = (__bf16*)(base + 3 * SZ_TOK);
    __bf16* h_b    = (__bf16*)(base);               // aliases n1/q/k/v (dead)
    __bf16* attn_b = (__bf16*)(base + 4 * SZ_TOK);
    __bf16* n2_b   = attn_b;                        // aliases attn (dead)
    float*  x1f    = (float*)(base + 5 * SZ_TOK);

    // weights -> bf16, swizzled into B-fragment order
    int nW = D_MODEL * D_MODEL, nF = D_MODEL * MLPD;
    cvtz_kernel<<<(nW + 255) / 256, 256, 0, stream>>>(wq, wq_b, D_MODEL, D_MODEL);
    cvtz_kernel<<<(nW + 255) / 256, 256, 0, stream>>>(wk, wk_b, D_MODEL, D_MODEL);
    cvtz_kernel<<<(nW + 255) / 256, 256, 0, stream>>>(wv, wv_b, D_MODEL, D_MODEL);
    cvtz_kernel<<<(nW + 255) / 256, 256, 0, stream>>>(wo, wo_b, D_MODEL, D_MODEL);
    cvtz_kernel<<<(nF + 255) / 256, 256, 0, stream>>>(wff1, wff1_b, D_MODEL, MLPD);
    cvtz_kernel<<<(nF + 255) / 256, 256, 0, stream>>>(wff2, wff2_b, MLPD, D_MODEL);

    // n1 = LN(x)
    ln_kernel<<<NTOK, 256, 0, stream>>>(x, g1, b1, n1_b);

    dim3 gD(D_MODEL / 64, NTOK / 64);   // 12 x 128
    dim3 gM(MLPD / 64, NTOK / 64);      // 48 x 128
    gemm_kernel<0, false, true><<<gD, 256, 0, stream>>>(n1_b, wq_b, bq, nullptr, q_b, NTOK, D_MODEL, D_MODEL);
    gemm_kernel<0, false, true><<<gD, 256, 0, stream>>>(n1_b, wk_b, bk, nullptr, k_b, NTOK, D_MODEL, D_MODEL);
    gemm_kernel<0, false, true><<<gD, 256, 0, stream>>>(n1_b, wv_b, bv, nullptr, v_b, NTOK, D_MODEL, D_MODEL);
    attn_kernel<<<dim3(BATCH * NHEAD, SEQ / 128), 256, 0, stream>>>(q_b, k_b, v_b, pb, attn_b);
    gemm_kernel<0, true, false><<<gD, 256, 0, stream>>>(attn_b, wo_b, bo, x, x1f, NTOK, D_MODEL, D_MODEL);
    ln_kernel<<<NTOK, 256, 0, stream>>>(x1f, g2, b2, n2_b);
    gemm_kernel<1, false, true><<<gM, 256, 0, stream>>>(n2_b, wff1_b, bff1, nullptr, h_b, NTOK, MLPD, D_MODEL);
    gemm_kernel<0, true, false><<<gD, 256, 0, stream>>>(h_b, wff2_b, bff2, x1f, d_out, NTOK, D_MODEL, MLPD);
}
